// LISA_36146444763331
// MI455X (gfx1250) — matrix operand order repaired and checked by reading, not compiled, not run on hardware
//
#include <hip/hip_runtime.h>
#include <hip/hip_bf16.h>

typedef _Float16 v16h __attribute__((ext_vector_type(16)));
typedef _Float16 v8h  __attribute__((ext_vector_type(8)));
typedef _Float16 v4h  __attribute__((ext_vector_type(4)));
typedef float    v8f  __attribute__((ext_vector_type(8)));

#define SHUF16 0, 1, 2, 3, 4, 5, 6, 7, 8, 9, 10, 11, 12, 13, 14, 15

namespace {
constexpr int kL         = 32768;   // points per batch row
constexpr int kNLat      = 2048;    // latent positions
constexpr int kDLat      = 64;      // latent channels
constexpr int kH         = 256;     // hidden width
constexpr int kThreads   = 512;     // 16 waves (wave32)
constexpr int kWaves     = kThreads / 32;
constexpr int kPPW       = kWaves * 16;            // 256 points per workgroup
constexpr int kActStride = 264;                    // halves per activation row (bank-rotating pad)
constexpr int kWHalves   = 8 * 16 * 512;           // 65536 halves = 128KB swizzled weight stage
constexpr int kActHalves = kWaves * 16 * kActStride;
constexpr size_t kSmemBytes = (size_t)(kWHalves + kActHalves) * sizeof(_Float16); // 266240 B
constexpr int kTotalPts  = 8 * kL;                 // 262144
}

// Stage one layer's f32 weights into LDS as f16, pre-swizzled into this kernel's WMMA
// B-fragment order: block (kt,nt) holds 32 lanes x 16 halves; lane L = col(n&15)
// + 16*(k>=16-half), split into two contiguous 16B runs so compute-side loads
// are single ds_load_b128 ops.
template <int NKT, bool FIRST>
__device__ __forceinline__ void load_weights(_Float16* __restrict__ smemW,
                                             const float* __restrict__ Wg) {
  for (int run = threadIdx.x; run < NKT * 512; run += kThreads) {
    const int kt    = run >> 9;
    const int rem   = run & 511;
    const int nt    = rem >> 5;
    const int ln    = rem & 31;
    const int n     = nt * 16 + (ln & 15);
    const int kbase = kt * 32 + ((ln >> 4) << 4);
    __align__(16) _Float16 tmp[16];
#pragma unroll
    for (int pos = 0; pos < 16; ++pos) {
      const int kd = kbase + pos;
      float v = 0.0f;
      if (FIRST) {
        // W0 rows remapped: input ch 0..12 -> rows 0..12, sampled ch -> rows 16..207,
        // rows 13..15 / 208..223 are zero padding (K0 padded to 224).
        if (kd < 13)                   v = Wg[kd * kH + n];
        else if (kd >= 16 && kd < 208) v = Wg[(kd - 3) * kH + n];
      } else {
        v = Wg[kd * kH + n];
      }
      tmp[pos] = (_Float16)v;
    }
    const int blk = (kt * 16 + nt) * 512;
    *(v8h*)&smemW[blk + ln * 8]       = *(const v8h*)&tmp[0];
    *(v8h*)&smemW[blk + 256 + ln * 8] = *(const v8h*)&tmp[8];
  }
}

// One 16xK @ KxH layer for this wave's 16 points: A fragments resident in VGPRs,
// B fragments streamed from the swizzled LDS stage, bias folded into the
// accumulator init, ReLU + f16 downconvert on the D writeback.
template <int NKT>
__device__ __forceinline__ void mlp_layer(const _Float16* __restrict__ smemW,
                                          _Float16* __restrict__ actW,
                                          const float* __restrict__ bias,
                                          int lane, int mrow, int ksel) {
  v16h A[NKT];
#pragma unroll
  for (int kt = 0; kt < NKT; ++kt) {
    // A layout used here (contiguous k, matches B stage): lane<16 holds row m K{0..15}; lane>=16 K{16..31}
    const v8h a0 = *(const v8h*)&actW[mrow * kActStride + kt * 32 + ksel * 16];
    const v8h a1 = *(const v8h*)&actW[mrow * kActStride + kt * 32 + ksel * 16 + 8];
    A[kt] = __builtin_shufflevector(a0, a1, SHUF16);
  }
  for (int nt = 0; nt < 16; ++nt) {
    const float bv = bias[nt * 16 + mrow];   // per-N bias == per-lane scalar in C layout
    v8f acc = {bv, bv, bv, bv, bv, bv, bv, bv};
#pragma unroll
    for (int kt = 0; kt < NKT; ++kt) {
      const int blk = (kt * 16 + nt) * 512;
      const v8h lo = *(const v8h*)&smemW[blk + lane * 8];
      const v8h hi = *(const v8h*)&smemW[blk + 256 + lane * 8];
      const v16h Bf = __builtin_shufflevector(lo, hi, SHUF16);
      acc = __builtin_amdgcn_wmma_f32_16x16x32_f16(false, A[kt], false, Bf,
                                                   (short)0, acc, false, false);
    }
#pragma unroll
    for (int r = 0; r < 8; ++r) {
      // D layout: VGPR r -> row r + 8*ksel, col nt*16 + mrow
      actW[(r + 8 * ksel) * kActStride + nt * 16 + mrow] =
          (_Float16)fmaxf(acc[r], 0.0f);
    }
  }
}

__global__ void __launch_bounds__(kThreads)
lisa_fused_mlp(const float* __restrict__ coord, const float* __restrict__ latent,
               const float* __restrict__ W0, const float* __restrict__ b0,
               const float* __restrict__ W1, const float* __restrict__ b1,
               const float* __restrict__ W2, const float* __restrict__ b2,
               const float* __restrict__ W3, const float* __restrict__ b3,
               const float* __restrict__ W4, const float* __restrict__ b4,
               float* __restrict__ out) {
  extern __shared__ __align__(16) _Float16 smem[];
  _Float16* smemW = smem;
  _Float16* actW  = smem + kWHalves + (threadIdx.x >> 5) * 16 * kActStride;

  const int lane   = threadIdx.x & 31;
  const int mrow   = lane & 15;
  const int ksel   = lane >> 4;
  const int ptBase = blockIdx.x * kPPW + (threadIdx.x >> 5) * 16;

  // ---- input features -> f16 activation rows in LDS ----
  if (lane < 16) {  // coord + positional encoding + zero padding, one point per lane
    const int gpt = ptBase + lane;
    const float c = coord[gpt];
    _Float16* row = &actW[lane * kActStride];
    row[0] = (_Float16)c;
#pragma unroll
    for (int j = 0; j < 6; ++j) {
      const float a = c * (float)(1 << j);
      row[1 + 2 * j] = (_Float16)__sinf(a);
      row[2 + 2 * j] = (_Float16)__cosf(a);
    }
    row[13] = (_Float16)0.f; row[14] = (_Float16)0.f; row[15] = (_Float16)0.f;
#pragma unroll
    for (int cc = 208; cc < 224; ++cc) row[cc] = (_Float16)0.f;
  }
  {  // border-clamped bilinear gather of unfolded latent: 2 lanes per point
    const int p    = lane >> 1;
    const int hs   = lane & 1;
    const int gpt  = ptBase + p;
    const int bidx = gpt >> 15;  // L = 32768
    const float c  = coord[gpt];
    const float ix = c * (float)kNLat - 0.5f;
    const float x0 = floorf(ix);
    const float t  = ix - x0;
    const float w0 = 1.0f - t;
    int i0 = (int)x0;       i0 = i0 < 0 ? 0 : (i0 > kNLat - 1 ? kNLat - 1 : i0);
    int i1 = (int)x0 + 1;   i1 = i1 < 0 ? 0 : (i1 > kNLat - 1 ? kNLat - 1 : i1);
    const float* base = latent + (size_t)bidx * kNLat * kDLat;
    const int r0[3] = {i0 > 0 ? i0 - 1 : 0, i0, i0 < kNLat - 1 ? i0 + 1 : kNLat - 1};
    const int r1[3] = {i1 > 0 ? i1 - 1 : 0, i1, i1 < kNLat - 1 ? i1 + 1 : kNLat - 1};
#pragma unroll
    for (int k = 0; k < 24; ++k) {
      const int cch = hs * 96 + k * 4;   // channel within [0,192): prev|self|next x 64
      const int g = cch >> 6;
      const int d = cch & 63;
      const float4 f0 = *(const float4*)(base + (size_t)r0[g] * kDLat + d);
      const float4 f1 = *(const float4*)(base + (size_t)r1[g] * kDLat + d);
      __align__(8) _Float16 hv[4];
      hv[0] = (_Float16)(w0 * f0.x + t * f1.x);
      hv[1] = (_Float16)(w0 * f0.y + t * f1.y);
      hv[2] = (_Float16)(w0 * f0.z + t * f1.z);
      hv[3] = (_Float16)(w0 * f0.w + t * f1.w);
      *(v4h*)&actW[p * kActStride + 16 + cch] = *(const v4h*)hv;  // cols 16..207
    }
  }

  // ---- 4 hidden layers, weights staged per layer ----
  load_weights<7, true>(smemW, W0);
  __syncthreads();
  mlp_layer<7>(smemW, actW, b0, lane, mrow, ksel);
  __syncthreads();

  load_weights<8, false>(smemW, W1);
  __syncthreads();
  mlp_layer<8>(smemW, actW, b1, lane, mrow, ksel);
  __syncthreads();

  load_weights<8, false>(smemW, W2);
  __syncthreads();
  mlp_layer<8>(smemW, actW, b2, lane, mrow, ksel);
  __syncthreads();

  load_weights<8, false>(smemW, W3);
  __syncthreads();
  mlp_layer<8>(smemW, actW, b3, lane, mrow, ksel);
  __syncthreads();

  // ---- final 256 -> 1 projection (VALU + cross-half shuffle reduce) ----
  float sum = 0.0f;
  const _Float16* hrow = &actW[mrow * kActStride + ksel * 128];
  const float* w4p = W4 + ksel * 128;
#pragma unroll 8
  for (int c = 0; c < 128; ++c) sum += (float)hrow[c] * w4p[c];
  sum += __shfl_xor(sum, 16, 32);
  if (lane < 16) out[ptBase + mrow] = sum + b4[0];
}

extern "C" void kernel_launch(void* const* d_in, const int* in_sizes, int n_in,
                              void* d_out, int out_size, void* d_ws, size_t ws_size,
                              hipStream_t stream) {
  (void)in_sizes; (void)n_in; (void)d_ws; (void)ws_size; (void)out_size;
  const float* coord  = (const float*)d_in[0];
  const float* latent = (const float*)d_in[1];
  const float* W0 = (const float*)d_in[2];
  const float* b0 = (const float*)d_in[3];
  const float* W1 = (const float*)d_in[4];
  const float* b1 = (const float*)d_in[5];
  const float* W2 = (const float*)d_in[6];
  const float* b2 = (const float*)d_in[7];
  const float* W3 = (const float*)d_in[8];
  const float* b3 = (const float*)d_in[9];
  const float* W4 = (const float*)d_in[10];
  const float* b4 = (const float*)d_in[11];
  float* out = (float*)d_out;

  dim3 grid(kTotalPts / kPPW);   // 1024 workgroups x 256 points
  lisa_fused_mlp<<<grid, kThreads, kSmemBytes, stream>>>(
      coord, latent, W0, b0, W1, b1, W2, b2, W3, b3, W4, b4, out);
}
